// HybridQLSTM_65481071396858
// MI455X (gfx1250) — matrix operand order repaired and checked by reading, not compiled, not run on hardware
//
#include <hip/hip_runtime.h>
#include <hip/hip_bf16.h>
#include <math.h>

// ---------------------------------------------------------------------------
// HybridQLSTM on MI455X (gfx1250), bf16 v_wmma_f32_16x16x32_bf16 path.
//
// All WMMA operands pre-packed in GLOBAL memory in the exact per-lane fragment
// layout (A: 16x32 MxK, B: 32x16 KxN per CDNA5 ISA). GEMM inner loops
// are pure global_load_b128 + v_wmma with an explicit distance-3 triple-buffer
// software pipeline pinned by sched_group_barrier (2x WMMA then 6x VMEM-read
// per unrolled iteration after an 18-load prologue) so both WMMAs of each
// iteration carry s_wait_loadcnt slack (round-4 asm: 0x4/0x0 -> target 0x8+).
//   - W1f/W2f : packed once at init (25 MB bf16, L2-resident, reused 256 steps)
//   - xf      : packed once at init for all T
//   - hf/pref : packed by the producer kernel's epilogue each step
// ---------------------------------------------------------------------------

typedef __bf16 bf16_t;
typedef __attribute__((ext_vector_type(16))) __bf16 v16bf;
typedef __attribute__((ext_vector_type(8)))  float  v8f;

#define T_STEPS 256
#define BATCH   64
#define DIM     1024
#define HID     1024
#define GDIM    (DIM + HID)   // 2048
#define FRAG    512           // bf16 elems per 32-lane fragment tile (32*16)

__device__ __forceinline__ float sigmoidf_(float x) {
    return 1.0f / (1.0f + __expf(-x));
}

#define WMMA_BF16(A, B, C) \
    __builtin_amdgcn_wmma_f32_16x16x32_bf16(false, (A), false, (B), (short)0, (C), false, false)

// Force issue pattern "2 WMMA then 6 VMEM loads" per unrolled iteration; the
// pipeline distance is set by the load prologue depth (3 iterations).
#if defined(__has_builtin)
#if __has_builtin(__builtin_amdgcn_sched_group_barrier)
#define SCHED_PIPE()                                          \
    do {                                                      \
        __builtin_amdgcn_sched_group_barrier(0x008, 2, 0);    \
        __builtin_amdgcn_sched_group_barrier(0x020, 6, 0);    \
    } while (0)
#endif
#endif
#ifndef SCHED_PIPE
#define SCHED_PIPE() do { } while (0)
#endif

// A-fragment (16x32 MxK): lane=(m&15)|(hi<<4), hi=(kk>>3)&1, j=(kk&7)|((kk&16)>>1)
// B-fragment (32x16 KxN): lane=(n&15)|(hi<<4), hi=(kk>>3)&1, j=(kk&7)|((kk&16)>>1)
// Packed linear: frag[tile][lane(32)][j(16)] -> 32 B contiguous per lane.

// ---------------------------------------------------------------------------
// init: pack W1 (fp32 [4,G,H]) -> B-fragment bf16 [4][16 ns][4 nsub][64 kc][32][16]
// ---------------------------------------------------------------------------
__global__ __launch_bounds__(256) void pack_w1(const float* __restrict__ W1,
                                               bf16_t* __restrict__ W1f) {
    size_t idx = (size_t)blockIdx.x * 256 + threadIdx.x;   // 8388608 total
    const int j    = idx & 15;
    const int lane = (idx >> 4) & 31;
    const int kc   = (idx >> 9) & 63;
    const int nsub = (idx >> 15) & 3;
    const int ns   = (idx >> 17) & 15;
    const int g    = (idx >> 21) & 3;
    const int kk = (j & 7) | ((lane & 16) >> 1) | ((j & 8) << 1);
    const int n  = ns * 64 + nsub * 16 + (lane & 15);
    const int k  = kc * 32 + kk;
    W1f[idx] = (bf16_t)W1[((size_t)g * GDIM + k) * HID + n];
}

// pack W2 (fp32 [4,H,H]) -> B-fragment bf16 [4][16 ns][4 nsub][32 kc][32][16]
__global__ __launch_bounds__(256) void pack_w2(const float* __restrict__ W2,
                                               bf16_t* __restrict__ W2f) {
    size_t idx = (size_t)blockIdx.x * 256 + threadIdx.x;   // 4194304 total
    const int j    = idx & 15;
    const int lane = (idx >> 4) & 31;
    const int kc   = (idx >> 9) & 31;
    const int nsub = (idx >> 14) & 3;
    const int ns   = (idx >> 16) & 15;
    const int g    = (idx >> 20) & 3;
    const int kk = (j & 7) | ((lane & 16) >> 1) | ((j & 8) << 1);
    const int n  = ns * 64 + nsub * 16 + (lane & 15);
    const int k  = kc * 32 + kk;
    W2f[idx] = (bf16_t)W2[((size_t)g * HID + k) * HID + n];
}

// pack x (fp32 [T,B,D]) -> A-fragment bf16 [T][4 mt][32 kc][32][16]
__global__ __launch_bounds__(256) void pack_x(const float* __restrict__ x,
                                              bf16_t* __restrict__ xf) {
    size_t idx = (size_t)blockIdx.x * 256 + threadIdx.x;   // 16777216 total
    const int j    = idx & 15;
    const int lane = (idx >> 4) & 31;
    const int kc   = (idx >> 9) & 31;
    const int mt   = (idx >> 14) & 3;
    const int t    = (int)(idx >> 16);
    const int hi = lane >> 4;
    const int kk = (j & 7) | (hi << 3) | ((j & 8) << 1);
    const int b  = mt * 16 + (lane & 15);
    const int k  = kc * 32 + kk;
    xf[idx] = (bf16_t)x[((size_t)t * BATCH + b) * DIM + k];
}

// zero c state (fp32) and h0 fragment (bf16)
__global__ __launch_bounds__(256) void init_state(float* __restrict__ cbuf,
                                                  bf16_t* __restrict__ hf) {
    int idx = blockIdx.x * 256 + threadIdx.x;              // 65536 total
    cbuf[idx] = 0.0f;
    hf[idx]   = (bf16_t)0.0f;
}

// ---------------------------------------------------------------------------
// gemm1: pre[g] = [x_t | h] . W1[g] + b1[g]      M=64, K=2048, N=64/block
// grid 64 = 4 gates x 16 n-slices; 8 waves; wave -> (m_tile, 2 n-subtiles).
// ---------------------------------------------------------------------------
__global__ __launch_bounds__(256) void lstm_gemm1(
    const bf16_t* __restrict__ xf,    // [T][4][32][32][16]
    const bf16_t* __restrict__ hf,    // [4][32][32][16]
    const bf16_t* __restrict__ W1f,   // [4][16][4][64][32][16]
    const float*  __restrict__ b1,    // [4,H]
    bf16_t* __restrict__ pref,        // [4][4][32][32][16]
    int t)
{
    const int lane = threadIdx.x & 31;
    const int wave = threadIdx.x >> 5;
    const int g  = blockIdx.x >> 4;
    const int ns = blockIdx.x & 15;
    const int mt = wave & 3;
    const int np = wave >> 2;                  // 0|1 -> n-subtiles {2np, 2np+1}

    const bf16_t* Ax = xf + ((((size_t)t * 4 + mt) * 32) * 32 + lane) * 16;
    const bf16_t* Ah = hf + (((size_t)mt * 32) * 32 + lane) * 16;
    const bf16_t* B0 = W1f + ((((size_t)(g * 16 + ns) * 4 + np * 2    ) * 64) * 32 + lane) * 16;
    const bf16_t* B1 = W1f + ((((size_t)(g * 16 + ns) * 4 + np * 2 + 1) * 64) * 32 + lane) * 16;

    // kc is compile-time under full unroll -> the x/h select folds away.
    auto loadA = [&](int kc) -> v16bf {
        return (kc < 32) ? *(const v16bf*)(Ax + (size_t)kc * FRAG)
                         : *(const v16bf*)(Ah + (size_t)(kc - 32) * FRAG);
    };

    __builtin_prefetch(B0, 0, 1);
    __builtin_prefetch(B1, 0, 1);

    v8f c0 = {};
    v8f c1 = {};

    // explicit triple-buffered pipeline, distance 3
    v16bf Abuf[3], Xbuf[3], Ybuf[3];
#pragma unroll
    for (int p = 0; p < 3; ++p) {
        Abuf[p] = loadA(p);
        Xbuf[p] = *(const v16bf*)(B0 + (size_t)p * FRAG);
        Ybuf[p] = *(const v16bf*)(B1 + (size_t)p * FRAG);
    }

#pragma unroll
    for (int kc = 0; kc < 64; ++kc) {
        const int cur = kc % 3;
        c0 = WMMA_BF16(Abuf[cur], Xbuf[cur], c0);
        c1 = WMMA_BF16(Abuf[cur], Ybuf[cur], c1);
        if (kc < 61) {
            Abuf[cur] = loadA(kc + 3);
            Xbuf[cur] = *(const v16bf*)(B0 + (size_t)(kc + 3) * FRAG);
            Ybuf[cur] = *(const v16bf*)(B1 + (size_t)(kc + 3) * FRAG);
            SCHED_PIPE();
        }
    }

    // epilogue: bias add, store pre as gemm2 A-fragments (scattered b16 stores)
    const int hi = lane >> 4;
#pragma unroll
    for (int s = 0; s < 2; ++s) {
        const v8f cc = s ? c1 : c0;
        const int n    = ns * 64 + (np * 2 + s) * 16 + (lane & 15);
        const float bias = b1[g * HID + n];
        const int kc2  = n >> 5;
        const int kk   = n & 31;
        const int hi2  = (kk >> 3) & 1;
        const int j2   = (kk & 7) | ((kk & 16) >> 1);
#pragma unroll
        for (int r = 0; r < 8; ++r) {
            const int bl    = r + 8 * hi;          // b & 15
            const int lane2 = bl | (hi2 << 4);
            pref[((((size_t)g * 4 + mt) * 32 + kc2) * 32 + lane2) * 16 + j2] =
                (bf16_t)(cc[r] + bias);
        }
    }
}

// ---------------------------------------------------------------------------
// gemm2 + cell: z[g] = pre[g] . W2[g] + b2[g]; then LSTM cell update.
// grid 16 = n-slices of H; each block owns a 64x64 (b,h) patch for ALL gates.
// ---------------------------------------------------------------------------
__global__ __launch_bounds__(256) void lstm_gemm2_cell(
    const bf16_t* __restrict__ pref,  // [4][4][32][32][16]
    const bf16_t* __restrict__ W2f,   // [4][16][4][32][32][16]
    const float*  __restrict__ b2,    // [4,H]
    float* __restrict__ cbuf,         // [B,H] fp32
    bf16_t* __restrict__ hf,          // [4][32][32][16] (next-step A)
    float* __restrict__ out,          // [T,B,H] ++ hx[B,H] ++ cx[B,H]
    int t)
{
    const int lane = threadIdx.x & 31;
    const int wave = threadIdx.x >> 5;
    const int ns = blockIdx.x;
    const int mt = wave & 3;
    const int np = wave >> 2;

    v8f acc[4][2];

#pragma unroll
    for (int g = 0; g < 4; ++g) {
        const bf16_t* Ag = pref + (((size_t)(g * 4 + mt) * 32) * 32 + lane) * 16;
        const bf16_t* B0 = W2f + ((((size_t)(g * 16 + ns) * 4 + np * 2    ) * 32) * 32 + lane) * 16;
        const bf16_t* B1 = W2f + ((((size_t)(g * 16 + ns) * 4 + np * 2 + 1) * 32) * 32 + lane) * 16;
        __builtin_prefetch(B0, 0, 1);

        v8f c0 = {};
        v8f c1 = {};

        // explicit triple-buffered pipeline, distance 3
        v16bf Abuf[3], Xbuf[3], Ybuf[3];
#pragma unroll
        for (int p = 0; p < 3; ++p) {
            Abuf[p] = *(const v16bf*)(Ag + (size_t)p * FRAG);
            Xbuf[p] = *(const v16bf*)(B0 + (size_t)p * FRAG);
            Ybuf[p] = *(const v16bf*)(B1 + (size_t)p * FRAG);
        }

#pragma unroll
        for (int kc = 0; kc < 32; ++kc) {
            const int cur = kc % 3;
            c0 = WMMA_BF16(Abuf[cur], Xbuf[cur], c0);
            c1 = WMMA_BF16(Abuf[cur], Ybuf[cur], c1);
            if (kc < 29) {
                Abuf[cur] = *(const v16bf*)(Ag + (size_t)(kc + 3) * FRAG);
                Xbuf[cur] = *(const v16bf*)(B0 + (size_t)(kc + 3) * FRAG);
                Ybuf[cur] = *(const v16bf*)(B1 + (size_t)(kc + 3) * FRAG);
                SCHED_PIPE();
            }
        }

        const float bias0 = b2[g * HID + ns * 64 + (np * 2    ) * 16 + (lane & 15)];
        const float bias1 = b2[g * HID + ns * 64 + (np * 2 + 1) * 16 + (lane & 15)];
#pragma unroll
        for (int r = 0; r < 8; ++r) { c0[r] += bias0; c1[r] += bias1; }
        acc[g][0] = c0;
        acc[g][1] = c1;
    }

    // cell update on this wave's C-layout elements
    const int hi = lane >> 4;
#pragma unroll
    for (int s = 0; s < 2; ++s) {
        const int n   = ns * 64 + (np * 2 + s) * 16 + (lane & 15);
        const int kc2 = n >> 5;
        const int kk  = n & 31;
        const int hi2 = (kk >> 3) & 1;
        const int j2  = (kk & 7) | ((kk & 16) >> 1);
#pragma unroll
        for (int r = 0; r < 8; ++r) {
            const int bl   = r + 8 * hi;                 // b & 15
            const int brow = mt * 16 + bl;
            const float f  = sigmoidf_(acc[0][s][r]);
            const float ii = sigmoidf_(acc[1][s][r]);
            const float gg = tanhf(acc[2][s][r]);
            const float oo = sigmoidf_(acc[3][s][r]);
            const size_t bh = (size_t)brow * HID + n;
            const float cnew = f * cbuf[bh] + ii * gg;
            const float hnew = oo * tanhf(cnew);
            cbuf[bh] = cnew;
            // h in A-fragment layout for next step's gemm1
            hf[(((size_t)mt * 32 + kc2) * 32 + (bl | (hi2 << 4))) * 16 + j2] = (bf16_t)hnew;
            out[(size_t)t * BATCH * HID + bh] = hnew;
            if (t == T_STEPS - 1) {
                out[(size_t)T_STEPS * BATCH * HID + bh] = hnew;                        // hx
                out[(size_t)T_STEPS * BATCH * HID + (size_t)BATCH * HID + bh] = cnew;  // cx
            }
        }
    }
}

// ---------------------------------------------------------------------------
extern "C" void kernel_launch(void* const* d_in, const int* in_sizes, int n_in,
                              void* d_out, int out_size, void* d_ws, size_t ws_size,
                              hipStream_t stream) {
    (void)in_sizes; (void)n_in; (void)out_size; (void)ws_size;

    const float* x  = (const float*)d_in[0];  // [T,B,D]
    const float* W1 = (const float*)d_in[1];  // [4,G,H]
    const float* b1 = (const float*)d_in[2];  // [4,H]
    const float* W2 = (const float*)d_in[3];  // [4,H,H]
    const float* b2 = (const float*)d_in[4];  // [4,H]
    float* out = (float*)d_out;

    // workspace carve-up (all 32B-aligned offsets)
    char* ws = (char*)d_ws;
    const size_t n_w1 = (size_t)4 * GDIM * HID;              // 8388608 bf16
    const size_t n_w2 = (size_t)4 * HID * HID;               // 4194304 bf16
    const size_t n_x  = (size_t)T_STEPS * BATCH * DIM;       // 16777216 bf16
    const size_t n_h  = (size_t)BATCH * HID;                 // 65536
    const size_t n_pre= (size_t)4 * BATCH * HID;             // 262144 bf16

    bf16_t* W1f = (bf16_t*)ws;
    bf16_t* W2f = W1f + n_w1;
    bf16_t* xf  = W2f + n_w2;
    bf16_t* hf  = xf + n_x;
    bf16_t* pref= hf + n_h;
    float*  cbuf= (float*)(pref + n_pre);

    pack_w1   <<<(int)(n_w1 / 256), 256, 0, stream>>>(W1, W1f);
    pack_w2   <<<(int)(n_w2 / 256), 256, 0, stream>>>(W2, W2f);
    pack_x    <<<(int)(n_x  / 256), 256, 0, stream>>>(x, xf);
    init_state<<<(int)(n_h  / 256), 256, 0, stream>>>(cbuf, hf);

    for (int t = 0; t < T_STEPS; ++t) {
        lstm_gemm1     <<<64, 256, 0, stream>>>(xf, hf, W1f, b1, pref, t);
        lstm_gemm2_cell<<<16, 256, 0, stream>>>(pref, W2f, b2, cbuf, hf, out, t);
    }
}
